// DotProductAttentionStream_17377437680478
// MI455X (gfx1250) — matrix operand order repaired and checked by reading, not compiled, not run on hardware
//
#include <hip/hip_runtime.h>
#include <hip/hip_bf16.h>

typedef __attribute__((ext_vector_type(16))) __bf16    v16bf;
typedef __attribute__((ext_vector_type(8)))  float     v8f;
typedef __attribute__((ext_vector_type(4)))  unsigned  v4u;
typedef __attribute__((ext_vector_type(8)))  unsigned  v8u;
typedef __attribute__((ext_vector_type(8)))  int       v8i;
typedef __attribute__((ext_vector_type(4)))  int       v4i;

#define B_SZ   16
#define N_SZ   2048
#define D_SZ   128
#define K_SEL  1536          // N * 3/4
#define WAVES  8
#define THREADS (WAVES * 32)

// LDS carve (dynamic shared):
//   [0, 131072)            : fp32 scores, 16 x 2048
//   [131072, 196864)       : bf16 P, 16 rows x 2056 (8-elem pad -> bank spread)
//   [196864, 205056)       : u32 hist, 8 waves x 256 bins
//   [205056, 205120)       : fp32 row scale, 16
//   [205120, 272704)       : TDM staging, 2 x (64 rows x 132 f32), padded rows
#define P_STRIDE   2056                 // bf16 elems per P row (4112B % 256 != 0)
#define P_OFF      131072
#define H_OFF      196864
#define SCL_OFF    205056
#define STG_OFF    205120
#define STG_STRIDE 132                  // 128 + 4-dword TDM pad -> bank spread
#define STG_HALF_B (64 * STG_STRIDE * 4)   // 33792 bytes per buffer
#define STG_HALF_F (64 * STG_STRIDE)       // floats per buffer
#define SMEM_BYTES (STG_OFF + 2 * STG_HALF_B)

// Order-preserving float -> uint map (monotone: a < b  <=>  ord(a) < ord(b))
__device__ __forceinline__ unsigned ordmap(float f) {
  unsigned u = __float_as_uint(f);
  return (u & 0x80000000u) ? ~u : (u | 0x80000000u);
}

// Tensor Data Mover: DMA a [rows x 128] fp32 tile (row stride 128 elements in
// memory) into LDS at byte offset lds_off, inserting a 4-dword pad every row
// (pad_interval=6 -> 512B, pad_amount=3 -> 4 dwords) so LDS rows stride 132.
// D# layout per CDNA5 ISA ch.8 (group0 + group1, 2D tile, groups 2/3 unused).
// This toolchain exposes the 6-arg builtin form (extra int32x8 group + cpol).
__device__ __forceinline__ void tdm_load_tile(unsigned lds_off, const float* gsrc,
                                              unsigned rows) {
  const unsigned long long ga = (unsigned long long)(uintptr_t)gsrc;
  v4u g0;
  g0[0] = 1u;                                        // count=1, user load D#
  g0[1] = lds_off;                                   // lds_addr (bytes)
  g0[2] = (unsigned)(ga & 0xffffffffu);              // global_addr[31:0]
  g0[3] = (unsigned)((ga >> 32) & 0x01ffffffu)       // global_addr[56:32]
          | (2u << 30);                              // type = 2 ("image")
  v8i g1;
  g1[0] = (int)((2u << 16)                           // data_size = 4 bytes
                | (1u << 20)                         // pad_enable
                | (6u << 22)                         // pad_interval: 512B
                | (3u << 25));                       // pad_amount: 4 dwords
  g1[1] = (int)((unsigned)(D_SZ & 0xffff) << 16);    // tensor_dim0 lo16
  g1[2] = (int)(((unsigned)D_SZ >> 16)               // tensor_dim0 hi16
                | ((unsigned)(N_SZ & 0xffff) << 16));// tensor_dim1 lo16
  g1[3] = (int)(((unsigned)N_SZ >> 16)               // tensor_dim1 hi16
                | ((unsigned)D_SZ << 16));           // tile_dim0 = 128 elems
  g1[4] = (int)(rows & 0xffffu);                     // tile_dim1 = rows, dim2=0
  g1[5] = (int)D_SZ;                                 // tensor_dim0_stride lo
  g1[6] = 0;                                         // stride0 hi | stride1 lo
  g1[7] = 0;                                         // stride1 hi
  const v4i z4 = {0, 0, 0, 0};
  const v8i z8 = {0, 0, 0, 0, 0, 0, 0, 0};
  __builtin_amdgcn_tensor_load_to_lds(g0, g1, z4, z4, z8, 0);
}

__global__ void __launch_bounds__(THREADS)
topk_attn_fused_17377437680478(const float* __restrict__ q,
                               const float* __restrict__ k,
                               const float* __restrict__ v,
                               float* __restrict__ out) {
  extern __shared__ char smem[];
  float*    sS   = (float*)smem;                     // 16 x 2048 f32 scores
  __bf16*   sP   = (__bf16*)(smem + P_OFF);          // 16 x 2056 bf16 probs
  unsigned* sH   = (unsigned*)(smem + H_OFF);        // 8 x 256 histogram
  float*    sScl = (float*)(smem + SCL_OFF);         // 16 row 1/sum
  float*    sStg = (float*)(smem + STG_OFF);         // TDM staging (2 buffers)

  const int tid  = threadIdx.x;
  const int lane = tid & 31;
  const int wave = tid >> 5;
  const int half = lane >> 4;   // lane group: 0 = lanes 0-15, 1 = lanes 16-31
  const int col  = lane & 15;

  const int b  = blockIdx.x >> 7;            // 128 row-tiles per batch
  const int m0 = (blockIdx.x & 127) << 4;    // first query row of this tile

  const float* qb = q + (size_t)b * N_SZ * D_SZ;
  const float* kb = k + (size_t)b * N_SZ * D_SZ;
  const float* vb = v + (size_t)b * N_SZ * D_SZ;

  // ---------------- Phase 1: Q tile -> 4 bf16 A-fragments (K-chunks of 32) --
  // A-layout (16-bit 16x32, ISA 7.12.2): lane holds row M = col; element e=2j+t
  // maps to K = (j>>2)*16 + half*8 + (j&3)*2 + t  => two 8-float runs per chunk.
  v16bf qa[4];
  {
    const float* qrow = qb + (size_t)(m0 + col) * D_SZ;
#pragma unroll
    for (int c = 0; c < 4; ++c) {
#pragma unroll
      for (int j = 0; j < 8; ++j) {
        const int d = 32 * c + ((j >> 2) << 4) + (half << 3) + ((j & 3) << 1);
        const float2 f2 = *(const float2*)(qrow + d);
        qa[c][2 * j]     = (__bf16)f2.x;
        qa[c][2 * j + 1] = (__bf16)f2.y;
      }
    }
  }

  // ---------------- Phase 2: S = Q K^T, waves stripe the 128 key tiles ------
  // Direct global loads: per-lane contiguous 32B runs, compiler pipelines the
  // b128 loads across key tiles with s_wait_loadcnt (no barriers needed).
  for (int t = wave; t < N_SZ / 16; t += WAVES) {
    const int k0 = t << 4;
    const float* krow = kb + (size_t)(k0 + col) * D_SZ;
    if (t + WAVES < N_SZ / 16)
      __builtin_prefetch(kb + (size_t)(k0 + (WAVES << 4) + col) * D_SZ, 0, 1);
    v8f acc = {};
#pragma unroll
    for (int c = 0; c < 4; ++c) {
      // B-layout (16-bit KxN): lane holds column N = col (the key), element e
      // maps to K-dim d = 32c + (e>>3)*16 + half*8 + (e&7)  => two 8-float runs.
      const float4* kp = (const float4*)(krow + 32 * c + (half << 3));
      v16bf bm;
#pragma unroll
      for (int e4 = 0; e4 < 4; ++e4) {
        const float4 f4 = kp[(e4 & 1) + ((e4 >> 1) << 2)];
        bm[4 * e4 + 0] = (__bf16)f4.x;
        bm[4 * e4 + 1] = (__bf16)f4.y;
        bm[4 * e4 + 2] = (__bf16)f4.z;
        bm[4 * e4 + 3] = (__bf16)f4.w;
      }
      acc = __builtin_amdgcn_wmma_f32_16x16x32_bf16(false, qa[c], false, bm,
                                                    (short)0, acc, false, false);
    }
    // D-layout: VGPR r, lane l -> (M = r + half*8, N = col)
#pragma unroll
    for (int r = 0; r < 8; ++r)
      sS[(size_t)(r + (half << 3)) * N_SZ + k0 + col] = acc[r];
  }
  __syncthreads();

  // ---------------- Phase 3: per-row radix select of k-th largest -----------
  // 2 rows per wave; 4 rounds of 8-bit MSB-first histogram select on ordmap().
  unsigned thrT[2];
  for (int rr = 0; rr < 2; ++rr) {
    const int row = wave * 2 + rr;
    const float* srow = sS + (size_t)row * N_SZ;
    unsigned prefix = 0, pmask = 0;
    int kneed = K_SEL;
    for (int shift = 24; shift >= 0; shift -= 8) {
#pragma unroll
      for (int i = 0; i < 8; ++i) sH[wave * 256 + lane + 32 * i] = 0;
      __syncthreads();
      for (int i = 0; i < N_SZ / 32; ++i) {
        const unsigned u = ordmap(srow[i * 32 + lane]);
        if ((u & pmask) == prefix)
          atomicAdd(&sH[wave * 256 + ((u >> shift) & 255u)], 1u);
      }
      __syncthreads();
      int selbin = 0, newk = kneed;
      if (lane == 0) {
        unsigned cum = 0;
        for (int bb = 255; bb >= 0; --bb) {
          const unsigned c = sH[wave * 256 + bb];
          if (cum + c >= (unsigned)kneed) { selbin = bb; newk = kneed - (int)cum; break; }
          cum += c;
        }
      }
      selbin = __shfl(selbin, 0);
      newk   = __shfl(newk, 0);
      prefix |= ((unsigned)selbin) << shift;
      pmask  |= 255u << shift;
      kneed = newk;
      __syncthreads();
    }
    thrT[rr] = prefix;   // ordered-uint value of the k-th largest score
  }

  // ---------------- Phase 4: thresholded softmax -> bf16 P (unnormalized) ---
  for (int rr = 0; rr < 2; ++rr) {
    const int row = wave * 2 + rr;
    const float* srow = sS + (size_t)row * N_SZ;
    const unsigned T = thrT[rr];
    float m = -3.402823466e+38f;
    for (int i = 0; i < N_SZ / 32; ++i) m = fmaxf(m, srow[i * 32 + lane]);
#pragma unroll
    for (int off = 16; off; off >>= 1) m = fmaxf(m, __shfl_xor(m, off));
    float ssum = 0.f;
    for (int i = 0; i < N_SZ / 32; ++i) {
      const float s = srow[i * 32 + lane];
      const float p = (ordmap(s) >= T) ? __expf(s - m) : 0.f;
      sP[(size_t)row * P_STRIDE + i * 32 + lane] = (__bf16)p;
      ssum += p;
    }
#pragma unroll
    for (int off = 16; off; off >>= 1) ssum += __shfl_xor(ssum, off);
    if (lane == 0) sScl[row] = 1.f / ssum;
  }
  __syncthreads();

  // ---------------- Phase 5: O = P V ; wave w owns D-chunk [16w, 16w+16) ----
  // Double-buffered TDM pipeline: 32 rounds of 64 keys. Wave 0 issues the DMA
  // for round r+1 into the other buffer, then s_wait_tensorcnt(1) so only the
  // current buffer must have landed (TENSORcnt completes in order per wave).
  const int d0 = wave << 4;
  v8f oacc = {};
  if (wave == 0) tdm_load_tile(STG_OFF, vb, 64);               // prologue
  for (int rnd = 0; rnd < 32; ++rnd) {
    const int kbase = rnd << 6;
    if (wave == 0) {
      if (rnd + 1 < 32) {
        tdm_load_tile(STG_OFF + (unsigned)((rnd + 1) & 1) * STG_HALF_B,
                      vb + (size_t)(kbase + 64) * D_SZ, 64);
        __builtin_amdgcn_s_wait_tensorcnt(1);
      } else {
        __builtin_amdgcn_s_wait_tensorcnt(0);
      }
    }
    __syncthreads();
    const float* buf = sStg + (size_t)(rnd & 1) * STG_HALF_F;
#pragma unroll
    for (int kk = 0; kk < 64; kk += 32) {
      const int k0 = kbase + kk;
      // A fragment from P: bf16 pairs are dword-aligned -> 8 ds_load_b32.
      v8u pw;
#pragma unroll
      for (int j = 0; j < 8; ++j) {
        const int koff = ((j >> 2) << 4) + (half << 3) + ((j & 3) << 1);
        pw[j] = *(const unsigned*)(sP + (size_t)col * P_STRIDE + k0 + koff);
      }
      const v16bf pa = __builtin_bit_cast(v16bf, pw);
      // B fragment from staged V (132-dword padded rows -> bank-spread).
      v16bf vm;
#pragma unroll
      for (int e = 0; e < 16; ++e) {
        const int keyl = kk + ((e >> 3) << 4) + (half << 3) + (e & 7);
        vm[e] = (__bf16)buf[(size_t)keyl * STG_STRIDE + d0 + col];
      }
      oacc = __builtin_amdgcn_wmma_f32_16x16x32_bf16(false, pa, false, vm,
                                                     (short)0, oacc, false, false);
    }
    __syncthreads();
  }
  float* ob = out + ((size_t)b * N_SZ + m0) * D_SZ;
#pragma unroll
  for (int r = 0; r < 8; ++r) {
    const int row = r + (half << 3);
    ob[(size_t)row * D_SZ + d0 + col] = oacc[r] * sScl[row];
  }
}

extern "C" void kernel_launch(void* const* d_in, const int* in_sizes, int n_in,
                              void* d_out, int out_size, void* d_ws, size_t ws_size,
                              hipStream_t stream) {
  (void)in_sizes; (void)n_in; (void)out_size; (void)d_ws; (void)ws_size;
  const float* q = (const float*)d_in[0];
  const float* k = (const float*)d_in[1];
  const float* v = (const float*)d_in[2];
  float* out = (float*)d_out;

  // ~266 KB of LDS per workgroup: legal on CDNA5 (320 KB/WGP) but above the
  // default cap, so raise the attribute (host API, graph-capture safe).
  (void)hipFuncSetAttribute((const void*)topk_attn_fused_17377437680478,
                            hipFuncAttributeMaxDynamicSharedMemorySize, SMEM_BYTES);

  dim3 grid(B_SZ * (N_SZ / 16));   // 2048 workgroups: (batch, 16-row tile)
  topk_attn_fused_17377437680478<<<grid, THREADS, SMEM_BYTES, stream>>>(q, k, v, out);
}
